// GeneratePrediction_9655086482259
// MI455X (gfx1250) — compile-verified, matrix operand order checked by reading, not run on hardware
//
#include <hip/hip_runtime.h>
#include <math.h>
#include <stdint.h>

// ---------------- problem constants ----------------
#define B_N   2048
#define C_N   650        // NUM_COMPO
#define V_N   27533      // NUM_CHARS
#define K_TOP 10
#define C_PAD 672        // K padded to multiple of 32 (650 -> 672)
#define N_PAD 656        // N padded to multiple of 16 (650 -> 656)
#define W_WORDS 861      // ceil(27533/32)
#define W_PAD   864      // padded word stride

// output layout (floats, concatenated in reference return order)
#define OFF_CIDX   0
#define OFF_CSCORE 20480
#define OFF_CS     40960
#define OFF_ADJ    1372160
#define OFF_HIT    2703360
#define OFF_HSCORE 2723840
#define OFF_P1     2725888
#define OFF_P2     2727936
#define OFF_P3     2729984

typedef __attribute__((ext_vector_type(16))) _Float16 v16h;
typedef __attribute__((ext_vector_type(8)))  float    v8f;

#define NEG_INF (-3.4e38f)

// ---------------- kernel 1: pack 0/1 matrix to bitmask ----------------
__global__ void pack_matrix_kernel(const int* __restrict__ mat, uint32_t* __restrict__ packed) {
    int c = blockIdx.x;
    for (int w = threadIdx.x; w < W_PAD; w += blockDim.x) {
        uint32_t bits = 0;
        int base = w * 32;
        if (w < W_WORDS) {
            int lim = V_N - base; if (lim > 32) lim = 32;
            const int* row = mat + (size_t)c * V_N + base;
            for (int j = 0; j < lim; ++j) bits |= (uint32_t)(row[j] & 1) << j;
        }
        packed[(size_t)c * W_PAD + w] = bits;
    }
}

// ---------------- kernel 2: sigmoid + thresholded f16 W ----------------
__global__ void compo_convert_kernel(const float* __restrict__ logits,
                                     float* __restrict__ out_cs,
                                     _Float16* __restrict__ Wf16) {
    int i = blockIdx.x * blockDim.x + threadIdx.x;
    if (i >= B_N * C_PAD) return;
    int b = i / C_PAD, c = i % C_PAD;
    _Float16 w = (_Float16)0.0f;
    if (c < C_N) {
        float s = 1.0f / (1.0f + expf(-logits[(size_t)b * C_N + c]));
        out_cs[(size_t)b * C_N + c] = s;
        if (s > 0.88f) w = (_Float16)s;
    }
    Wf16[(size_t)b * C_PAD + c] = w;
}

// ---------------- kernel 3: co_occ -> f16 transposed [N_PAD][C_PAD] ----------------
__global__ void cooc_convert_kernel(const float* __restrict__ co, _Float16* __restrict__ coT) {
    int i = blockIdx.x * blockDim.x + threadIdx.x;
    if (i >= N_PAD * C_PAD) return;
    int n = i / C_PAD, k = i % C_PAD;
    _Float16 v = (_Float16)0.0f;
    if (n < C_N && k < C_N) v = (_Float16)co[(size_t)k * C_N + n];
    coT[(size_t)n * C_PAD + k] = v;
}

// ---------------- kernel 4: WMMA GEMM  adjusted = cs + 0.1 * (W @ co_occ) ----------------
union Frag { v16h h; uint32_t u[8]; };

__global__ void __launch_bounds__(32)
adjust_gemm_kernel(const _Float16* __restrict__ Wf16,
                   const _Float16* __restrict__ coT,
                   const float* __restrict__ cs,
                   float* __restrict__ adj) {
    int tile_m = blockIdx.x;          // 0..127
    int tile_n = blockIdx.y;          // 0..40
    int lane   = threadIdx.x;         // wave32
    int mn     = lane & 15;
    int khalf  = lane >> 4;

    const _Float16* arow = Wf16 + (size_t)(tile_m * 16 + mn) * C_PAD;
    const _Float16* brow = coT  + (size_t)(tile_n * 16 + mn) * C_PAD;

    v8f acc = {};
    for (int kb = 0; kb < C_PAD; kb += 32) {
        Frag a, bfrag;
#pragma unroll
        for (int h2 = 0; h2 < 8; ++h2) {
            // ISA 16-bit A layout: half h -> K = (h&7) + 8*khalf + (h>=8 ? 16 : 0)
            int k = kb + 2 * (h2 & 3) + 8 * khalf + ((h2 >= 4) ? 16 : 0);
            a.u[h2]     = *(const uint32_t*)(arow + k);
            bfrag.u[h2] = *(const uint32_t*)(brow + k);
        }
        acc = __builtin_amdgcn_wmma_f32_16x16x32_f16(false, a.h, false, bfrag.h,
                                                     (short)0, acc, false, false);
    }

    int col = tile_n * 16 + mn;
    if (col < C_N) {
#pragma unroll
        for (int r = 0; r < 8; ++r) {
            int row = tile_m * 16 + r + 8 * khalf;   // C/D layout: lanes 16-31 hold M=r+8
            size_t o = (size_t)row * C_N + col;
            adj[o] = cs[o] + 0.1f * acc[r];
        }
    }
}

// ---------------- kernel 5: single-pass online softmax + top-10 per class row ----------------
__global__ void __launch_bounds__(256)
class_topk_kernel(const float* __restrict__ logits,
                  float* __restrict__ out_idx, float* __restrict__ out_score) {
    int b = blockIdx.x;
    int t = threadIdx.x;
    const float* row = logits + (size_t)b * V_N;

    __shared__ float redm[256];   // running max (pair reduction)
    __shared__ float reds[256];   // running sum (pair reduction)
    __shared__ float red[256];
    __shared__ int   redi[256];
    __shared__ float tv[256 * K_TOP];
    __shared__ int   ti[256 * K_TOP];
    __shared__ int   thead[256];

    // single pass: online (max, sum) + per-thread top-10 of logits (registers)
    float lv[K_TOP]; int li[K_TOP];
#pragma unroll
    for (int k = 0; k < K_TOP; ++k) { lv[k] = NEG_INF; li[k] = 0; }

    float m = NEG_INF, s = 0.0f;
    for (int i = t; i < V_N; i += 256) {
        float x = row[i];
        if (x > m) { s = s * expf(m - x) + 1.0f; m = x; }
        else       { s += expf(x - m); }
        if (x > lv[K_TOP - 1]) {
            lv[K_TOP - 1] = x; li[K_TOP - 1] = i;
#pragma unroll
            for (int p = K_TOP - 1; p > 0; --p) {
                if (lv[p] > lv[p - 1]) {
                    float fv = lv[p]; lv[p] = lv[p - 1]; lv[p - 1] = fv;
                    int   iv = li[p]; li[p] = li[p - 1]; li[p - 1] = iv;
                }
            }
        }
    }

    // combine (max, sum) pairs across the block
    redm[t] = m; reds[t] = s; __syncthreads();
    for (int st = 128; st > 0; st >>= 1) {
        if (t < st) {
            float ma = redm[t], mb = redm[t + st];
            float M  = fmaxf(ma, mb);
            reds[t]  = reds[t] * expf(ma - M) + reds[t + st] * expf(mb - M);
            redm[t]  = M;
        }
        __syncthreads();
    }
    float rowmax    = redm[0];
    float inv_denom = 1.0f / reds[0];
    __syncthreads();

    // merge 256 sorted lists: 10 rounds of block argmax over list heads
#pragma unroll
    for (int k = 0; k < K_TOP; ++k) { tv[t * K_TOP + k] = lv[k]; ti[t * K_TOP + k] = li[k]; }
    thead[t] = 0;
    __syncthreads();

    for (int k = 0; k < K_TOP; ++k) {
        red[t]  = tv[t * K_TOP + thead[t]];
        redi[t] = t;
        __syncthreads();
        for (int st = 128; st > 0; st >>= 1) {
            if (t < st) {
                float va = red[t], vb = red[t + st];
                int wa = redi[t], wb = redi[t + st];
                int ia = ti[wa * K_TOP + thead[wa]];
                int ib = ti[wb * K_TOP + thead[wb]];
                if (vb > va || (vb == va && ib < ia)) { red[t] = vb; redi[t] = wb; }
            }
            __syncthreads();
        }
        if (t == 0) {
            int w = redi[0];
            int src = w * K_TOP + thead[w];
            out_idx[(size_t)b * K_TOP + k]   = (float)ti[src];
            out_score[(size_t)b * K_TOP + k] = expf(tv[src] - rowmax) * inv_denom;
            thead[w]++;
        }
        __syncthreads();
    }
}

// ---------------- block helpers (blockDim.x == 256, uniform call sites) ----------------
__device__ __forceinline__ int block_sum(int v, int* redi, int t) {
    redi[t] = v; __syncthreads();
    for (int s = 128; s > 0; s >>= 1) { if (t < s) redi[t] += redi[t + s]; __syncthreads(); }
    int r = redi[0]; __syncthreads();
    return r;
}

__device__ __forceinline__ void block_argmax(float v, int idx, float* redf, int* redi, int t,
                                             float* outv, int* outi) {
    redf[t] = v; redi[t] = idx; __syncthreads();
    for (int s = 128; s > 0; s >>= 1) {
        if (t < s) {
            float vb = redf[t + s];
            if (vb > redf[t] || (vb == redf[t] && redi[t + s] < redi[t])) {
                redf[t] = vb; redi[t] = redi[t + s];
            }
        }
        __syncthreads();
    }
    *outv = redf[0]; *outi = redi[0]; __syncthreads();
}

// ---------------- kernel 6: per-sample decision ----------------
__global__ void __launch_bounds__(256)
decide_kernel(const float* __restrict__ cs_all,
              const float* __restrict__ adj_all,
              const uint32_t* __restrict__ packed,
              const float* __restrict__ cls_idx,
              const float* __restrict__ cls_score,
              float* __restrict__ out_hit,
              float* __restrict__ out_hscore,
              float* __restrict__ out_p1,
              float* __restrict__ out_p2,
              float* __restrict__ out_p3) {
    int b = blockIdx.x, t = threadIdx.x;

    __shared__ float scores[C_N];
    __shared__ float avail[C_N];
    __shared__ uint32_t hit[W_PAD];
    __shared__ uint32_t prev[W_PAD];
    __shared__ float redf[256];
    __shared__ int   redi[256];

    const float* cs  = cs_all  + (size_t)b * C_N;
    const float* adj = adj_all + (size_t)b * C_N;

    // load scores, build avail mask, local argmax(adjusted), local num_sel
    float lm = NEG_INF; int lmi = 0; int lsel = 0;
    for (int c = t; c < C_N; c += 256) {
        float s = cs[c];
        scores[c] = s;
        avail[c]  = (s > 0.7f) ? s : NEG_INF;
        if (s > 0.7f) lsel++;
        float a = adj[c];
        if (a > lm) { lm = a; lmi = c; }
    }
    __syncthreads();

    float dummyv; int max_idx;
    block_argmax(lm, lmi, redf, redi, t, &dummyv, &max_idx);
    int num_sel = block_sum(lsel, redi, t);
    float max_score = scores[max_idx];

    // hit = matrix[max_idx] (bitset); prev = hit
    const uint32_t* prow = packed + (size_t)max_idx * W_PAD;
    int lc = 0;
    for (int w = t; w < W_PAD; w += 256) {
        uint32_t x = prow[w];
        hit[w] = x; prev[w] = x;
        lc += __popc(x);
    }
    __syncthreads();
    int hitcnt = block_sum(lc, redi, t);

    // pop order[0] (top selected score) -- reference loop starts at i=1
    float sA = max_score, sB = max_score;
    {
        float bv = NEG_INF; int bi = 0;
        for (int c = t; c < C_N; c += 256) { float v = avail[c]; if (v > bv) { bv = v; bi = c; } }
        float v0; int c0;
        block_argmax(bv, bi, redf, redi, t, &v0, &c0);
        if (t == 0) avail[c0] = NEG_INF;
        __syncthreads();
        sA = v0;
    }

    int i = 1;
    while (i < num_sel && hitcnt > 1) {
        // pop order[i]
        float bv = NEG_INF; int bi = 0;
        for (int c = t; c < C_N; c += 256) { float v = avail[c]; if (v > bv) { bv = v; bi = c; } }
        float v; int cidx;
        block_argmax(bv, bi, redf, redi, t, &v, &cidx);
        if (t == 0) avail[cidx] = NEG_INF;
        __syncthreads();

        // intersect bitsets
        const uint32_t* crow = packed + (size_t)cidx * W_PAD;
        int lcc = 0;
        for (int w = t; w < W_PAD; w += 256) {
            uint32_t h = hit[w];
            prev[w] = h;
            uint32_t nh = h & crow[w];
            hit[w] = nh;
            lcc += __popc(nh);
        }
        __syncthreads();
        hitcnt = block_sum(lcc, redi, t);

        sB = sA; sA = v; ++i;
    }

    if (hitcnt == 0) {  // rollback to previous intersection
        --i;
        int lcc = 0;
        for (int w = t; w < W_PAD; w += 256) { uint32_t h = prev[w]; hit[w] = h; lcc += __popc(h); }
        __syncthreads();
        hitcnt = block_sum(lcc, redi, t);
        sA = sB;
    }
    float hit_score = (i == 1) ? max_score : sA;

    // serial epilogue: extract first K set bits, predictions
    if (t == 0) {
        int hidx[K_TOP];
        int cnt = 0;
        for (int w = 0; w < W_WORDS && cnt < K_TOP; ++w) {
            uint32_t x = hit[w];
            while (x && cnt < K_TOP) {
                int bpos = __ffs(x) - 1;
                hidx[cnt++] = w * 32 + bpos;
                x &= x - 1;
            }
        }
        for (int k = 0; k < K_TOP; ++k)
            out_hit[(size_t)b * K_TOP + k] = (k < cnt) ? (float)hidx[k] : -1.0f;
        out_hscore[b] = hit_score;

        int ci[K_TOP];
        for (int k = 0; k < K_TOP; ++k) ci[k] = (int)cls_idx[(size_t)b * K_TOP + k];
        float cs0 = cls_score[(size_t)b * K_TOP];
        int num_hit = cnt;
        out_p1[b] = (num_hit == 1) ? (float)hidx[0] : (float)ci[0];
        out_p2[b] = (cs0 < 0.85f && num_hit == 1) ? (float)hidx[0] : (float)ci[0];
        int j = 0; bool found = false;
        for (int k = 0; k < K_TOP && !found; ++k)
            for (int h = 0; h < cnt; ++h)
                if (ci[k] == hidx[h]) { j = k; found = true; break; }
        out_p3[b] = (float)ci[j];
    }
}

// ---------------- launch ----------------
extern "C" void kernel_launch(void* const* d_in, const int* in_sizes, int n_in,
                              void* d_out, int out_size, void* d_ws, size_t ws_size,
                              hipStream_t stream) {
    (void)in_sizes; (void)n_in; (void)out_size; (void)ws_size;

    const float* class_logits = (const float*)d_in[0];
    const float* compo_logits = (const float*)d_in[1];
    /* d_in[2] chinese_char_ids: unused by reference outputs */
    const float* co_occ       = (const float*)d_in[3];
    const int*   matrix       = (const int*)d_in[4];
    float* out = (float*)d_out;

    char* ws = (char*)d_ws;
    _Float16* Wf16   = (_Float16*)ws;                              // 2048*672*2   = 2,752,512 B
    _Float16* coT    = (_Float16*)(ws + 2752512);                  //  656*672*2   =   881,664 B
    uint32_t* packed = (uint32_t*)(ws + 3634176);                  //  650*864*4   = 2,246,400 B

    float* o_cidx   = out + OFF_CIDX;
    float* o_cscore = out + OFF_CSCORE;
    float* o_cs     = out + OFF_CS;
    float* o_adj    = out + OFF_ADJ;
    float* o_hit    = out + OFF_HIT;
    float* o_hscore = out + OFF_HSCORE;
    float* o_p1     = out + OFF_P1;
    float* o_p2     = out + OFF_P2;
    float* o_p3     = out + OFF_P3;

    pack_matrix_kernel<<<C_N, 256, 0, stream>>>(matrix, packed);

    int n1 = B_N * C_PAD;
    compo_convert_kernel<<<(n1 + 255) / 256, 256, 0, stream>>>(compo_logits, o_cs, Wf16);

    int n2 = N_PAD * C_PAD;
    cooc_convert_kernel<<<(n2 + 255) / 256, 256, 0, stream>>>(co_occ, coT);

    dim3 gemm_grid(B_N / 16, N_PAD / 16);
    adjust_gemm_kernel<<<gemm_grid, 32, 0, stream>>>(Wf16, coT, o_cs, o_adj);

    class_topk_kernel<<<B_N, 256, 0, stream>>>(class_logits, o_cidx, o_cscore);

    decide_kernel<<<B_N, 256, 0, stream>>>(o_cs, o_adj, packed, o_cidx, o_cscore,
                                           o_hit, o_hscore, o_p1, o_p2, o_p3);
}
